// HybridKernel_65481071409756
// MI455X (gfx1250) — matrix operand order repaired and checked by reading, not compiled, not run on hardware
//
#include <hip/hip_runtime.h>

// ---------------------------------------------------------------------------
// RBF kernel matrix: out = exp(-gamma * max(||x||^2 + ||y||^2 - 2 x y^T, 0))
// N = M = 8192, D = 256, fp32 in/out.
//
// Pass 1 (once): per-row squared norms + fp32 -> (bf16 hi, bf16 lo) split
//                into d_ws (removes redundant conversion VALU from the GEMM
//                and reads hi+lo = 4 B/element, same bytes as fp32).
// Pass 2: 128x256 block tile, 8 wave32 waves, 64x64 per wave (acc[4][4]),
//         v_wmma_f32_16x16x32_bf16 with 3-product hi/lo accumulation,
//         register-pipelined global->LDS staging, fused exp epilogue.
// ---------------------------------------------------------------------------

typedef __bf16 bf16_t;
typedef __attribute__((ext_vector_type(16))) __bf16 v16bf;
typedef __attribute__((ext_vector_type(8)))  float  v8f;

#define DIMS   256
#define NROWS  8192
#define MROWS  8192
#define BM     128
#define BN     256
#define BK     32
#define KSTEPS (DIMS / BK)   // 8
#define LDS_LD 40            // 32 + 8 bf16 pad -> 80B pitch (16B aligned, bank-friendly)

// ---- Pass 1: row squared norms + bf16 hi/lo split --------------------------
__global__ __launch_bounds__(256) void rbf_prep(const float* __restrict__ v,
                                                float* __restrict__ norms,
                                                bf16_t* __restrict__ hi,
                                                bf16_t* __restrict__ lo) {
  const int row  = blockIdx.x * 8 + (threadIdx.x >> 5);  // one wave32 per row
  const int lane = threadIdx.x & 31;
  const size_t base = (size_t)row * DIMS + lane * 8;     // 8 contiguous floats/lane
  float4 f0 = ((const float4*)(v + base))[0];
  float4 f1 = ((const float4*)(v + base))[1];
  float f[8] = {f0.x, f0.y, f0.z, f0.w, f1.x, f1.y, f1.z, f1.w};
  alignas(16) bf16_t h[8], l[8];
  float s = 0.f;
#pragma unroll
  for (int c = 0; c < 8; ++c) {
    s += f[c] * f[c];
    bf16_t hv = (bf16_t)f[c];
    h[c] = hv;
    l[c] = (bf16_t)(f[c] - (float)hv);
  }
  *(uint4*)(hi + base) = *(const uint4*)h;
  *(uint4*)(lo + base) = *(const uint4*)l;
#pragma unroll
  for (int off = 16; off > 0; off >>= 1) s += __shfl_xor(s, off, 32);
  if (lane == 0) norms[row] = s;
}

// ---- Pass 2: tiled WMMA GEMM + fused RBF epilogue --------------------------
__global__ __launch_bounds__(256) void rbf_wmma(const bf16_t* __restrict__ xhi,
                                                const bf16_t* __restrict__ xlo,
                                                const bf16_t* __restrict__ yhi,
                                                const bf16_t* __restrict__ ylo,
                                                const float* __restrict__ xn,
                                                const float* __restrict__ yn,
                                                const float* __restrict__ gamma_p,
                                                float* __restrict__ out) {
  __shared__ bf16_t Ah[BM * LDS_LD];   // 10 KB
  __shared__ bf16_t Al[BM * LDS_LD];   // 10 KB
  __shared__ bf16_t Bh[BN * LDS_LD];   // 20 KB
  __shared__ bf16_t Bl[BN * LDS_LD];   // 20 KB
  __shared__ float  sXn[BM];
  __shared__ float  sYn[BN];

  const int tid = threadIdx.x;
  const int i0  = blockIdx.y * BM;     // x-row tile base
  const int j0  = blockIdx.x * BN;     // y-row tile base

  if (tid < BM) sXn[tid] = xn[i0 + tid];
  sYn[tid] = yn[j0 + tid];

  const int wave = tid >> 5;
  const int lane = tid & 31;
  const int wmm  = wave >> 2;          // 0..1 : 64-row band
  const int wmn  = wave & 3;           // 0..3 : 64-col band
  const int fr   = lane & 15;
  const int half = lane >> 4;

  // Staging assignment (copy-only; data is pre-split bf16):
  //   A: tid<128 copies hi row tid, tid>=128 copies lo row tid-128 (32 bf16 = 64B)
  //   B: every tid copies hi row tid and lo row tid
  const int arow = tid & (BM - 1);
  const bf16_t* asrc  = (tid < BM ? xhi : xlo) + (size_t)(i0 + arow) * DIMS;
  bf16_t*       adst  = (tid < BM ? Ah : Al) + arow * LDS_LD;
  const bf16_t* bhsrc = yhi + (size_t)(j0 + tid) * DIMS;
  const bf16_t* blsrc = ylo + (size_t)(j0 + tid) * DIMS;
  bf16_t*       bhdst = Bh + tid * LDS_LD;
  bf16_t*       bldst = Bl + tid * LDS_LD;

  uint4 ra[4], rbh[4], rbl[4];         // register pipeline: next k-step's panels
  auto load_regs = [&](int k0) {
#pragma unroll
    for (int c = 0; c < 4; ++c) {
      ra[c]  = ((const uint4*)(asrc + k0))[c];
      rbh[c] = ((const uint4*)(bhsrc + k0))[c];
      rbl[c] = ((const uint4*)(blsrc + k0))[c];
    }
  };
  auto store_lds = [&]() {
#pragma unroll
    for (int c = 0; c < 4; ++c) {
      *(uint4*)(adst + 8 * c)  = ra[c];
      *(uint4*)(bhdst + 8 * c) = rbh[c];
      *(uint4*)(bldst + 8 * c) = rbl[c];
    }
  };

  v8f acc[4][4];
  const v8f vzero = {0.f, 0.f, 0.f, 0.f, 0.f, 0.f, 0.f, 0.f};
#pragma unroll
  for (int a = 0; a < 4; ++a)
#pragma unroll
    for (int b = 0; b < 4; ++b) acc[a][b] = vzero;

  load_regs(0);
  for (int ks = 0; ks < KSTEPS; ++ks) {
    store_lds();
    __syncthreads();                       // staged panels visible
    if (ks + 1 < KSTEPS) load_regs((ks + 1) * BK);   // overlap with WMMA below

    // A fragments: row = lane&15, K chunks at half*8 and 16+half*8
    v16bf ah[4], al[4];
#pragma unroll
    for (int mt = 0; mt < 4; ++mt) {
      const int rbase = (wmm * 64 + mt * 16 + fr) * LDS_LD;
      ((uint4*)&ah[mt])[0] = *(const uint4*)&Ah[rbase + half * 8];
      ((uint4*)&ah[mt])[1] = *(const uint4*)&Ah[rbase + 16 + half * 8];
      ((uint4*)&al[mt])[0] = *(const uint4*)&Al[rbase + half * 8];
      ((uint4*)&al[mt])[1] = *(const uint4*)&Al[rbase + 16 + half * 8];
    }
#pragma unroll
    for (int nt = 0; nt < 4; ++nt) {
      // B fragment: col = lane&15, K chunks at half*8 and 16+half*8
      const int cbase = (wmn * 64 + nt * 16 + fr) * LDS_LD;
      v16bf bh, bl;
      ((uint4*)&bh)[0] = *(const uint4*)&Bh[cbase + half * 8];
      ((uint4*)&bh)[1] = *(const uint4*)&Bh[cbase + 16 + half * 8];
      ((uint4*)&bl)[0] = *(const uint4*)&Bl[cbase + half * 8];
      ((uint4*)&bl)[1] = *(const uint4*)&Bl[cbase + 16 + half * 8];
#pragma unroll
      for (int mt = 0; mt < 4; ++mt) {
        acc[mt][nt] = __builtin_amdgcn_wmma_f32_16x16x32_bf16(
            false, al[mt], false, bh, (short)0, acc[mt][nt], false, false);
        acc[mt][nt] = __builtin_amdgcn_wmma_f32_16x16x32_bf16(
            false, ah[mt], false, bl, (short)0, acc[mt][nt], false, false);
        acc[mt][nt] = __builtin_amdgcn_wmma_f32_16x16x32_bf16(
            false, ah[mt], false, bh, (short)0, acc[mt][nt], false, false);
      }
    }
    __syncthreads();                       // LDS consumed; next store_lds is safe
  }

  // ---- epilogue: d2 = xn + yn - 2*dot, out = exp(-gamma * max(d2, 0)) ----
  const float g = *gamma_p;
#pragma unroll
  for (int mt = 0; mt < 4; ++mt) {
    const int lm = wmm * 64 + mt * 16 + half * 8;   // local row base of 8 acc rows
#pragma unroll
    for (int nt = 0; nt < 4; ++nt) {
      const int ln  = wmn * 64 + nt * 16 + fr;      // local col
      const float ynv = sYn[ln];
      float* po = out + (size_t)(i0 + lm) * MROWS + (j0 + ln);
#pragma unroll
      for (int rr = 0; rr < 8; ++rr) {
        float d2 = fmaxf(sXn[lm + rr] + ynv - 2.f * acc[mt][nt][rr], 0.f);
        po[(size_t)rr * MROWS] = __expf(-g * d2);
      }
    }
  }
}

// ---------------------------------------------------------------------------
extern "C" void kernel_launch(void* const* d_in, const int* in_sizes, int n_in,
                              void* d_out, int out_size, void* d_ws, size_t ws_size,
                              hipStream_t stream) {
  const float* x     = (const float*)d_in[0];
  const float* y     = (const float*)d_in[1];
  const float* gamma = (const float*)d_in[2];
  float* out = (float*)d_out;

  // d_ws layout: xn | yn | xhi | xlo | yhi | ylo  (~16.06 MB)
  float* xn = (float*)d_ws;
  float* yn = xn + NROWS;
  bf16_t* xhi = (bf16_t*)(yn + MROWS);
  bf16_t* xlo = xhi + (size_t)NROWS * DIMS;
  bf16_t* yhi = xlo + (size_t)NROWS * DIMS;
  bf16_t* ylo = yhi + (size_t)MROWS * DIMS;

  rbf_prep<<<NROWS / 8, 256, 0, stream>>>(x, xn, xhi, xlo);
  rbf_prep<<<MROWS / 8, 256, 0, stream>>>(y, yn, yhi, ylo);

  dim3 grid(MROWS / BN, NROWS / BM);   // (32, 64)
  rbf_wmma<<<grid, 256, 0, stream>>>(xhi, xlo, yhi, ylo, xn, yn, gamma, out);
}
